// UltraLight_CrackNet_50483045597307
// MI455X (gfx1250) — compile-verified, matrix operand order checked by reading, not run on hardware
//
#include <hip/hip_runtime.h>
#include <hip/hip_bf16.h>

// ---------------- problem constants (from reference) ----------------
#define BB   4
#define HH   64
#define WW   64
#define DMOD 96     // DM
#define NST  16     // DS (state dim)
#define DI_  192    // EXPAND*DM
#define LL   4096   // H*W
#define KDIR 4      // K scan directions
#define HID_ 768    // 4*DI
#define NROW 38     // DTR + 2*DS
#define NDTR 6      // ceil(DM/16)

// ---------------- WMMA / TDM types ----------------
typedef __bf16 bf16_t;
typedef bf16_t v16bf __attribute__((ext_vector_type(16)));
typedef float v8f __attribute__((ext_vector_type(8)));
typedef unsigned int  v4u   __attribute__((ext_vector_type(4)));
typedef unsigned int  u32x4 __attribute__((ext_vector_type(4)));
typedef int           i32x4 __attribute__((ext_vector_type(4)));
typedef int           i32x8 __attribute__((ext_vector_type(8)));

union Frag { v4u q[2]; v16bf bf; };   // 32B: two b128 loads = one 16x32 bf16 fragment

#if defined(__has_builtin)
#  if __has_builtin(__builtin_amdgcn_tensor_load_to_lds) && __has_builtin(__builtin_amdgcn_s_wait_tensorcnt)
#    define USE_TDM 1
#  endif
#endif
#ifndef USE_TDM
#  define USE_TDM 0
#endif

__device__ __forceinline__ unsigned short f2bfu(float f) {
    union { float f; unsigned u; } v; v.f = f;
    unsigned r = v.u + 0x7FFFu + ((v.u >> 16) & 1u);   // round-to-nearest-even
    return (unsigned short)(r >> 16);
}

__device__ __forceinline__ float silu_f(float x) {
    return x / (1.0f + __expf(-x));
}

// ---------------- f32 -> packed bf16 (2 elems/thread, b32 stores) ----------------
__global__ void pack_bf16(const float* __restrict__ src, unsigned short* __restrict__ dst, int n) {
    int i = blockIdx.x * blockDim.x + threadIdx.x;     // pair index
    if (2 * i >= n) return;
    unsigned lo = f2bfu(src[2 * i]);
    unsigned hi = f2bfu(src[2 * i + 1]);
    ((unsigned*)dst)[i] = lo | (hi << 16);
}

#if USE_TDM
// TDM: stage one contiguous 16 x KD bf16 tile (rows of A) into LDS.
// D# per ISA ch.8: group0 = {count|flags, lds_addr, gaddr_lo, gaddr_hi|type=2},
// group1 packs data_size=2B, tensor 16 x KD, tile 16 x KD, stride KD.
__device__ __forceinline__ void tdm_load_tile_bf16(const unsigned short* gsrc, void* ldst, int KD) {
    unsigned long long ga = (unsigned long long)(uintptr_t)gsrc;
    unsigned lds = (unsigned)(uintptr_t)ldst;          // low 32 bits = LDS byte offset
    u32x4 g0;
    g0[0] = 1u;                                        // count=1, user descriptor
    g0[1] = lds;                                       // lds_addr
    g0[2] = (unsigned)ga;                              // global_addr[31:0]
    g0[3] = (unsigned)((ga >> 32) & 0x01FFFFFFu) | (2u << 30);  // addr[56:32] | type=2
    i32x8 g1;
    g1[0] = (int)(1u << 16);                           // wg_mask=0 | data_size=1 (2 bytes)
    g1[1] = (int)(((unsigned)KD & 0xFFFFu) << 16);     // tensor_dim0[15:0]
    g1[2] = (int)(16u << 16);                          // tensor_dim0 hi | tensor_dim1=16
    g1[3] = (int)((unsigned)KD << 16);                 // tensor_dim1 hi | tile_dim0=KD
    g1[4] = 16;                                        // tile_dim1=16 | tile_dim2=0
    g1[5] = KD;                                        // tensor_dim0_stride lo32
    g1[6] = 0;
    g1[7] = 0;
    i32x4 z4 = {0, 0, 0, 0};
#if __clang_major__ >= 23
    i32x8 z8 = {0, 0, 0, 0, 0, 0, 0, 0};
    __builtin_amdgcn_tensor_load_to_lds(g0, g1, z4, z4, z8, 0);
#else
    __builtin_amdgcn_tensor_load_to_lds(g0, g1, z4, z4, 0);
#endif
}
#endif

// ============================================================================
// WMMA GEMM on pre-packed bf16:  C[m,n] = epi( sum_k A[m,k]*Wt[n,k] + bias[n] )
// A: (M,KD) bf16.  Wt: (N,KD) bf16 (weight; B = Wt^T).  KD compile-time ->
// fully unrolled K loop (KD/32 static v_wmma per instantiation).
// A tile (16 x KD) staged in LDS by TDM (or cooperative copy fallback);
// fragments read via ds_load_b128; B fragments via global_load_b128.
// ============================================================================
template <int KD, int ACT, int ADD>
__global__ void gemm_bf16_wmma(const unsigned short* __restrict__ A,
                               const unsigned short* __restrict__ Wt,
                               const float* __restrict__ bias,
                               float* __restrict__ C,
                               int N,
                               const float* __restrict__ addsrc,
                               const float* __restrict__ skipp) {
    __shared__ __align__(16) unsigned short sA[16 * KD];
    const int lane  = threadIdx.x & 31;
    const int wave  = threadIdx.x >> 5;
    const int mBase = blockIdx.x * 16;
    const int nBase = blockIdx.y * 128 + wave * 16;

    // ---- stage A tile (all 16 rows x KD, contiguous in global) ----
#if USE_TDM
    if (wave == 0) {
        tdm_load_tile_bf16(A + (size_t)mBase * KD, sA, KD);
        __builtin_amdgcn_s_wait_tensorcnt(0);
    }
    __syncthreads();
#else
    {
        const unsigned* gsrc = (const unsigned*)(A + (size_t)mBase * KD);
        unsigned* d = (unsigned*)sA;
        for (int i = threadIdx.x; i < 16 * KD / 2; i += 256) d[i] = gsrc[i];
        __syncthreads();
    }
#endif

    if (nBase >= N) return;                    // wave-uniform; after barrier

    const int r16 = lane & 15;                 // A row / B col within tile
    const int kb  = (lane >> 4) * 8;           // lane-half K base (0 or 8)
    const unsigned short* __restrict__ aptr = sA + r16 * KD;
    const unsigned short* __restrict__ brow = Wt + (size_t)(nBase + r16) * KD;

    v8f acc = {};
#pragma unroll
    for (int k0 = 0; k0 < KD; k0 += 32) {
        Frag fa, fb;
        fa.q[0] = *(const v4u*)(aptr + k0 + kb);        // K = kb..kb+7   (ds_load_b128)
        fa.q[1] = *(const v4u*)(aptr + k0 + 16 + kb);   // K = 16+kb..+7
        fb.q[0] = *(const v4u*)(brow + k0 + kb);        // global_load_b128
        fb.q[1] = *(const v4u*)(brow + k0 + 16 + kb);
        acc = __builtin_amdgcn_wmma_f32_16x16x32_bf16(
                  false, fa.bf, false, fb.bf, (short)0, acc, false, false);
    }

    // D layout: lane 0-15 -> N=lane, M=v; lane 16-31 -> N=lane-16, M=v+8
    const int col   = nBase + r16;
    const int rbase = mBase + ((lane >> 4) ? 8 : 0);
    if (col < N) {
        const float bv = bias ? bias[col] : 0.0f;
        const float sk = ADD ? skipp[0] : 1.0f;
#pragma unroll
        for (int v = 0; v < 8; ++v) {
            float r = acc[v] + bv;
            if (ACT == 1) r = silu_f(r);
            if (ADD) r = addsrc[(size_t)(rbase + v) * N + col] + sk * r;
            C[(size_t)(rbase + v) * N + col] = r;
        }
    }
}

// ============================================================================
// Depthwise 3x3 conv + bias + SiLU.  xi lives in xz (B,L,2*DI) cols [0,DI).
// Output xc in (B,DI,L) "NCHW" layout for the scan stage.
// ============================================================================
__global__ void dwconv3_silu(const float* __restrict__ xz,
                             const float* __restrict__ cw,
                             const float* __restrict__ cb,
                             float* __restrict__ xc) {
    int idx = blockIdx.x * blockDim.x + threadIdx.x;   // (b,d,l)
    if (idx >= BB * DI_ * LL) return;
    int l = idx % LL; int t = idx / LL;
    int d = t % DI_;  int b = t / DI_;
    int h = l / WW, w = l % WW;
    float acc = cb[d];
#pragma unroll
    for (int dh = -1; dh <= 1; ++dh)
#pragma unroll
        for (int dw = -1; dw <= 1; ++dw) {
            int hh = h + dh, ww = w + dw;
            if (hh < 0 || hh >= HH || ww < 0 || ww >= WW) continue;
            float xv = xz[((size_t)b * LL + hh * WW + ww) * (2 * DI_) + d];
            acc += xv * cw[d * 9 + (dh + 1) * 3 + (dw + 1)];
        }
    xc[((size_t)b * DI_ + d) * LL + l] = silu_f(acc);
}

// direction-k source index into the (row-major) xc L dimension
__device__ __forceinline__ int xs_src(int k, int l) {
    int ll = (k & 2) ? (LL - 1 - l) : l;
    if (k & 1) { int h = ll % HH; int w = ll / HH; return h * WW + w; }
    return ll;
}

// ============================================================================
// Scan prep: x_dbl = x_proj_w[k] @ xs  (38x192 per (b,k,l)), then
// delta = softplus(dt_projs_w @ dts + dt_b), plus Bs/Cs extraction.
// Projection matrices resident in LDS (uniform broadcast reads).
// ============================================================================
__global__ void scan_prep(const float* __restrict__ xc,
                          const float* __restrict__ xpw,
                          const float* __restrict__ dtw,
                          const float* __restrict__ dtb,
                          float* __restrict__ delta,
                          float* __restrict__ Bs,
                          float* __restrict__ Cs) {
    __shared__ float s_xp[NROW * DI_];   // 38*192
    __shared__ float s_dw[DI_ * NDTR];   // 192*6
    __shared__ float s_db[DI_];
    const int bk = blockIdx.x;           // b*4+k
    const int b  = bk >> 2, k = bk & 3;
    for (int i = threadIdx.x; i < NROW * DI_; i += blockDim.x) s_xp[i] = xpw[k * NROW * DI_ + i];
    for (int i = threadIdx.x; i < DI_ * NDTR; i += blockDim.x) s_dw[i] = dtw[k * DI_ * NDTR + i];
    for (int i = threadIdx.x; i < DI_; i += blockDim.x)        s_db[i] = dtb[k * DI_ + i];
    __syncthreads();

    const int l   = blockIdx.y * blockDim.x + threadIdx.x;
    const int src = xs_src(k, l);
    float acc[NROW];
#pragma unroll
    for (int c = 0; c < NROW; ++c) acc[c] = 0.f;
    const float* __restrict__ xcb = xc + (size_t)b * DI_ * LL;
    for (int d = 0; d < DI_; ++d) {
        float xv = xcb[(size_t)d * LL + src];
#pragma unroll
        for (int c = 0; c < NROW; ++c) acc[c] += s_xp[c * DI_ + d] * xv;
    }
#pragma unroll
    for (int n = 0; n < NST; ++n) {
        Bs[((size_t)bk * NST + n) * LL + l] = acc[NDTR + n];
        Cs[((size_t)bk * NST + n) * LL + l] = acc[NDTR + NST + n];
    }
    for (int d = 0; d < DI_; ++d) {
        float s = s_db[d];
#pragma unroll
        for (int r = 0; r < NDTR; ++r) s += s_dw[d * NDTR + r] * acc[r];
        float dl = (s > 20.f) ? s : log1pf(__expf(s));   // softplus
        delta[((size_t)bk * DI_ + d) * LL + l] = dl;
    }
}

// ============================================================================
// Selective scan: serial over L, parallel over B*K*DI channels.
// One block per (b,k); thread d carries 16-state recurrence in registers.
// B_t/C_t staged through LDS (broadcast to all channel threads).
// ============================================================================
__global__ void selective_scan(const float* __restrict__ xc,
                               const float* __restrict__ delta,
                               const float* __restrict__ Bs,
                               const float* __restrict__ Cs,
                               const float* __restrict__ A_logs,
                               const float* __restrict__ Dsv,
                               float* __restrict__ ys) {
    __shared__ float sB[NST], sC[NST];
    const int bk = blockIdx.x;
    const int b  = bk >> 2, k = bk & 3;
    const int d  = threadIdx.x;
    float a[NST], h[NST];
#pragma unroll
    for (int n = 0; n < NST; ++n) {
        a[n] = -__expf(A_logs[((size_t)(k * DI_ + d)) * NST + n]);
        h[n] = 0.f;
    }
    const float Dv = Dsv[k * DI_ + d];
    const float* __restrict__ xcb = xc + (size_t)b * DI_ * LL + (size_t)d * LL;
    const float* __restrict__ dlt = delta + ((size_t)bk * DI_ + d) * LL;
    const float* __restrict__ Bp  = Bs + (size_t)bk * NST * LL;
    const float* __restrict__ Cp  = Cs + (size_t)bk * NST * LL;
    float* __restrict__ yp = ys + ((size_t)bk * DI_ + d) * LL;

    for (int l = 0; l < LL; ++l) {
        if (threadIdx.x < NST)            sB[threadIdx.x]       = Bp[threadIdx.x * LL + l];
        else if (threadIdx.x < 2 * NST)   sC[threadIdx.x - NST] = Cp[(threadIdx.x - NST) * LL + l];
        __syncthreads();
        const float xv = xcb[xs_src(k, l)];
        const float dl = dlt[l];
        const float dx = dl * xv;
        float y = 0.f;
#pragma unroll
        for (int n = 0; n < NST; ++n) {
            h[n] = h[n] * __expf(dl * a[n]) + dx * sB[n];
            y += h[n] * sC[n];
        }
        yp[l] = y + Dv * xv;
        __syncthreads();
    }
}

// ---------------- wave32 block reduction ----------------
template <int NWAVE>
__device__ __forceinline__ float block_sum(float v, float* sred) {
#pragma unroll
    for (int off = 16; off > 0; off >>= 1) v += __shfl_xor(v, off, 32);
    const int wave = threadIdx.x >> 5, lane = threadIdx.x & 31;
    if (lane == 0) sred[wave] = v;
    __syncthreads();
    if (threadIdx.x == 0) {
        float t = 0.f;
#pragma unroll
        for (int i = 0; i < NWAVE; ++i) t += sred[i];
        sred[0] = t;
    }
    __syncthreads();
    float r = sred[0];
    __syncthreads();
    return r;
}

// ============================================================================
// Combine 4 directions (with flips/transposes), LayerNorm over DI,
// gate with SiLU(z).  One block per (b,l), thread per channel d.
// ============================================================================
__global__ void combine_ln_gate(const float* __restrict__ ys,
                                const float* __restrict__ xz,
                                const float* __restrict__ g,
                                const float* __restrict__ beta,
                                float* __restrict__ yg) {
    __shared__ float sred[8];
    const int bl = blockIdx.x;
    const int b  = bl / LL, l = bl % LL;
    const int d  = threadIdx.x;
    const int h  = l / WW, w = l % WW;
    const int lwh = w * HH + h;
    const size_t sb = (size_t)b * KDIR * DI_ * LL;
    float v = ys[sb + ((size_t)(0 * DI_ + d)) * LL + l]
            + ys[sb + ((size_t)(2 * DI_ + d)) * LL + (LL - 1 - l)]
            + ys[sb + ((size_t)(1 * DI_ + d)) * LL + lwh]
            + ys[sb + ((size_t)(3 * DI_ + d)) * LL + (LL - 1 - lwh)];

    const float m  = block_sum<6>(v, sred) * (1.0f / DI_);
    const float df = v - m;
    const float var = block_sum<6>(df * df, sred) * (1.0f / DI_);
    const float yn = df * rsqrtf(var + 1e-5f) * g[d] + beta[d];
    const float z  = xz[((size_t)b * LL + l) * (2 * DI_) + DI_ + d];
    yg[((size_t)b * LL + l) * DI_ + d] = yn * silu_f(z);
}

// ============================================================================
// FFN multiscale depthwise conv: hc = h1 + dw1*h1 + dw3(*)h1 + dw5(*)h1
// ============================================================================
__global__ void msconv(const float* __restrict__ h1,
                       const float* __restrict__ w1,
                       const float* __restrict__ w3,
                       const float* __restrict__ w5,
                       float* __restrict__ hc) {
    int idx = blockIdx.x * blockDim.x + threadIdx.x;   // (b,l,c), c fastest
    if (idx >= BB * LL * HID_) return;
    const int c  = idx % HID_;
    const int bl = idx / HID_;
    const int l  = bl % LL, b = bl / LL;
    const int h  = l / WW, w = l % WW;
    const float center = h1[idx];
    float acc = center + w1[c] * center;
#pragma unroll
    for (int dh = -1; dh <= 1; ++dh)
#pragma unroll
        for (int dw = -1; dw <= 1; ++dw) {
            int hh = h + dh, ww = w + dw;
            if (hh < 0 || hh >= HH || ww < 0 || ww >= WW) continue;
            acc += w3[c * 9 + (dh + 1) * 3 + (dw + 1)]
                 * h1[((size_t)b * LL + hh * WW + ww) * HID_ + c];
        }
#pragma unroll
    for (int dh = -2; dh <= 2; ++dh)
#pragma unroll
        for (int dw = -2; dw <= 2; ++dw) {
            int hh = h + dh, ww = w + dw;
            if (hh < 0 || hh >= HH || ww < 0 || ww >= WW) continue;
            acc += w5[c * 25 + (dh + 2) * 5 + (dw + 2)]
                 * h1[((size_t)b * LL + hh * WW + ww) * HID_ + c];
        }
    hc[idx] = acc;
}

// ============================================================================
// LayerNorm over HID=768; block 256 threads, 3 channels per thread.
// ============================================================================
__global__ void ln768(const float* __restrict__ hc,
                      const float* __restrict__ g,
                      const float* __restrict__ beta,
                      float* __restrict__ hn) {
    __shared__ float sred[8];
    const int bl = blockIdx.x;
    const float* __restrict__ row = hc + (size_t)bl * HID_;
    float lv[3], s = 0.f;
#pragma unroll
    for (int i = 0; i < 3; ++i) { lv[i] = row[threadIdx.x + i * 256]; s += lv[i]; }
    const float m = block_sum<8>(s, sred) * (1.0f / HID_);
    float vs = 0.f;
#pragma unroll
    for (int i = 0; i < 3; ++i) { float df = lv[i] - m; vs += df * df; }
    const float inv = rsqrtf(block_sum<8>(vs, sred) * (1.0f / HID_) + 1e-5f);
#pragma unroll
    for (int i = 0; i < 3; ++i) {
        int c = threadIdx.x + i * 256;
        hn[(size_t)bl * HID_ + c] = (lv[i] - m) * inv * g[c] + beta[c];
    }
}

// ============================================================================
extern "C" void kernel_launch(void* const* d_in, const int* in_sizes, int n_in,
                              void* d_out, int out_size, void* d_ws, size_t ws_size,
                              hipStream_t stream) {
    const float* x      = (const float*)d_in[0];
    const float* in_w   = (const float*)d_in[1];
    const float* conv_w = (const float*)d_in[2];
    const float* conv_b = (const float*)d_in[3];
    const float* xpw    = (const float*)d_in[4];
    const float* dtw    = (const float*)d_in[5];
    const float* dtb    = (const float*)d_in[6];
    const float* A_logs = (const float*)d_in[7];
    const float* Dsv    = (const float*)d_in[8];
    const float* ong    = (const float*)d_in[9];
    const float* onb    = (const float*)d_in[10];
    const float* opw    = (const float*)d_in[11];
    const float* fiw    = (const float*)d_in[12];
    const float* fib    = (const float*)d_in[13];
    const float* dw1    = (const float*)d_in[14];
    const float* dw3    = (const float*)d_in[15];
    const float* dw5    = (const float*)d_in[16];
    const float* flg    = (const float*)d_in[17];
    const float* flb    = (const float*)d_in[18];
    const float* fow    = (const float*)d_in[19];
    const float* fob    = (const float*)d_in[20];
    const float* skip   = (const float*)d_in[21];
    float* out = (float*)d_out;
    float* ws  = (float*)d_ws;

    // ---- f32 workspace layout, with dead-buffer reuse ----
    float* xz    = ws;                                   // (B,L,2*DI)
    float* xc    = xz    + (size_t)BB * LL * 2 * DI_;    // (B,DI,L)
    float* delta = xc    + (size_t)BB * DI_ * LL;        // (B,K,DI,L)
    float* Bsb   = delta + (size_t)BB * KDIR * DI_ * LL; // (B,K,16,L)
    float* Csb   = Bsb   + (size_t)BB * KDIR * NST * LL; // (B,K,16,L)
    float* ys    = Csb   + (size_t)BB * KDIR * NST * LL; // (B,K,DI,L)
    float* yg    = ys    + (size_t)BB * KDIR * DI_ * LL; // (B,L,DI)
    float* outp  = yg    + (size_t)BB * LL * DI_;        // (B,L,DM)
    float* h1    = outp  + (size_t)BB * LL * DMOD;       // (B,L,HID)
    float* fend  = h1    + (size_t)BB * LL * HID_;
    float* hc    = ys;     // reuse: ys dead after combine_ln_gate
    float* hn    = delta;  // reuse: delta dead after selective_scan

    // ---- bf16 workspace (packed activations + weights) ----
    unsigned short* bfb    = (unsigned short*)fend;
    unsigned short* x_bf   = bfb;                                    // 16384*96
    unsigned short* inw_bf = x_bf   + (size_t)BB * LL * DMOD;        // 384*96
    unsigned short* opw_bf = inw_bf + (size_t)2 * DI_ * DMOD;        // 96*192
    unsigned short* fiw_bf = opw_bf + (size_t)DMOD * DI_;            // 768*192
    unsigned short* fow_bf = fiw_bf + (size_t)HID_ * DI_;            // 96*768
    unsigned short* yg_bf  = fow_bf + (size_t)DMOD * HID_;           // 16384*192
    unsigned short* hn_bf  = yg_bf  + (size_t)BB * LL * DI_;         // 16384*768

    const int M = BB * LL;  // 16384 tokens
    auto pkg = [](int n) { return (n / 2 + 255) / 256; };

    // 0) pack weights + input to bf16
    pack_bf16<<<pkg(M * DMOD), 256, 0, stream>>>(x, x_bf, M * DMOD);
    pack_bf16<<<pkg(2 * DI_ * DMOD), 256, 0, stream>>>(in_w, inw_bf, 2 * DI_ * DMOD);
    pack_bf16<<<pkg(DMOD * DI_), 256, 0, stream>>>(opw, opw_bf, DMOD * DI_);
    pack_bf16<<<pkg(HID_ * DI_), 256, 0, stream>>>(fiw, fiw_bf, HID_ * DI_);
    pack_bf16<<<pkg(DMOD * HID_), 256, 0, stream>>>(fow, fow_bf, DMOD * HID_);

    // 1) xz = x @ in_proj_w^T                       (WMMA bf16, K=96, 3 wmma)
    gemm_bf16_wmma<DMOD, 0, 0><<<dim3(M / 16, 3), 256, 0, stream>>>(
        x_bf, inw_bf, nullptr, xz, 2 * DI_, nullptr, nullptr);
    // 2) xc = silu(dwconv3(xi) + b)
    dwconv3_silu<<<(BB * DI_ * LL + 255) / 256, 256, 0, stream>>>(xz, conv_w, conv_b, xc);
    // 3) x_dbl / dt_proj / softplus  -> delta, Bs, Cs
    scan_prep<<<dim3(BB * KDIR, LL / 128), 128, 0, stream>>>(xc, xpw, dtw, dtb, delta, Bsb, Csb);
    // 4) selective scan (serial in L, parallel across 3072 channels)
    selective_scan<<<BB * KDIR, DI_, 0, stream>>>(xc, delta, Bsb, Csb, A_logs, Dsv, ys);
    // 5) 4-direction combine + LayerNorm(DI) + SiLU(z) gate -> yg (B,L,DI)
    combine_ln_gate<<<BB * LL, DI_, 0, stream>>>(ys, xz, ong, onb, yg);
    // 5b) pack yg
    pack_bf16<<<pkg(M * DI_), 256, 0, stream>>>(yg, yg_bf, M * DI_);
    // 6) outp = yg @ out_proj_w^T                   (WMMA, K=192, 6 wmma)
    gemm_bf16_wmma<DI_, 0, 0><<<dim3(M / 16, 1), 256, 0, stream>>>(
        yg_bf, opw_bf, nullptr, outp, DMOD, nullptr, nullptr);
    // 7) h1 = silu(yg @ ffn_in_w^T + b)             (WMMA, K=192, 6 wmma)
    gemm_bf16_wmma<DI_, 1, 0><<<dim3(M / 16, 6), 256, 0, stream>>>(
        yg_bf, fiw_bf, fib, h1, HID_, nullptr, nullptr);
    // 8) hc = h1 + dw1*h1 + dw3(*)h1 + dw5(*)h1
    msconv<<<(BB * LL * HID_ + 255) / 256, 256, 0, stream>>>(h1, dw1, dw3, dw5, hc);
    // 9) hn = LayerNorm(hc) over HID
    ln768<<<BB * LL, 256, 0, stream>>>(hc, flg, flb, hn);
    // 9b) pack hn
    pack_bf16<<<pkg(M * HID_), 256, 0, stream>>>(hn, hn_bf, M * HID_);
    // 10) out = outp + skip * (hn @ ffn_out_w^T + b)  (WMMA, K=768, 24 wmma, fused epilogue)
    gemm_bf16_wmma<HID_, 0, 1><<<dim3(M / 16, 1), 256, 0, stream>>>(
        hn_bf, fow_bf, fob, out, DMOD, outp, skip);
}
